// CVNet_SG_66090956751198
// MI455X (gfx1250) — matrix operand order repaired and checked by reading, not compiled, not run on hardware
//
#include <hip/hip_runtime.h>
#include <hip/hip_bf16.h>

#define N_ANCH   8400
#define NUM_CLS  600
#define H_IMG    640
#define W_IMG    640
#define PRE_NMS  1024
#define M_MAX    31
#define OUT_SZ   224
#define NP_SORT  16384   // next pow2 >= 8400

typedef __attribute__((ext_vector_type(16))) _Float16 v16h;
typedef __attribute__((ext_vector_type(8)))  float    v8f;

__device__ __forceinline__ unsigned float_key(float f) {
    unsigned u = __float_as_uint(f);
    return (u & 0x80000000u) ? ~u : (u | 0x80000000u);
}

// ---------------------------------------------------------------------------
// K1: decode boxes + max/argmax over 600 class logits.
//     One thread = 4 anchors -> global_load_b128 on the dominant 20MB stream.
// ---------------------------------------------------------------------------
__global__ __launch_bounds__(256) void k_decode(const float* __restrict__ raw,
                                                float* __restrict__ scores,
                                                int*   __restrict__ classes,
                                                float* __restrict__ boxes) {
    int t = blockIdx.x * blockDim.x + threadIdx.x;   // quad index
    if (t >= N_ANCH / 4) return;
    int a0 = t * 4;

    float4 cx = *(const float4*)(raw + 0 * N_ANCH + a0);
    float4 cy = *(const float4*)(raw + 1 * N_ANCH + a0);
    float4 w  = *(const float4*)(raw + 2 * N_ANCH + a0);
    float4 h  = *(const float4*)(raw + 3 * N_ANCH + a0);
    const float* pcx = &cx.x; const float* pcy = &cy.x;
    const float* pw  = &w.x;  const float* ph  = &h.x;
    #pragma unroll
    for (int q = 0; q < 4; ++q) {
        boxes[(a0 + q) * 4 + 0] = pcx[q] - 0.5f * pw[q];
        boxes[(a0 + q) * 4 + 1] = pcy[q] - 0.5f * ph[q];
        boxes[(a0 + q) * 4 + 2] = pcx[q] + 0.5f * pw[q];
        boxes[(a0 + q) * 4 + 3] = pcy[q] + 0.5f * ph[q];
    }

    float4 bv = *(const float4*)(raw + 4 * N_ANCH + a0);
    float best[4] = { bv.x, bv.y, bv.z, bv.w };
    int   bi[4]   = { 0, 0, 0, 0 };
    for (int c = 1; c < NUM_CLS; ++c) {
        const float* p = raw + (size_t)(4 + c) * N_ANCH + a0;
        __builtin_prefetch(p + 8 * N_ANCH, 0, 0);    // global_prefetch_b8
        float4 v = *(const float4*)p;
        const float* pv = &v.x;
        #pragma unroll
        for (int q = 0; q < 4; ++q)
            if (pv[q] > best[q]) { best[q] = pv[q]; bi[q] = c; }  // first-max wins
    }
    *(float4*)(scores + a0) = make_float4(best[0], best[1], best[2], best[3]);
    *(int4*)(classes + a0)  = make_int4(bi[0], bi[1], bi[2], bi[3]);
}

// ---------------------------------------------------------------------------
// K2: exact top-1024 via single-WG bitonic sort of 16384 u64 keys in LDS
//     (128 KB dynamic LDS; CDNA5 WGP has 320 KB). Key=(fbits<<32)|~idx.
// ---------------------------------------------------------------------------
__global__ __launch_bounds__(1024) void k_sort_topk(const float* __restrict__ scores,
                                                    const int*   __restrict__ classes,
                                                    const float* __restrict__ boxes,
                                                    float* __restrict__ b_out,
                                                    float* __restrict__ sc_out,
                                                    int*   __restrict__ cl_out) {
    extern __shared__ unsigned long long skeys[];   // NP_SORT entries
    int t = threadIdx.x;
    for (int i = t; i < NP_SORT; i += 1024) {
        unsigned long long k = 0ull;
        if (i < N_ANCH) {
            unsigned u = float_key(scores[i]);
            k = ((unsigned long long)u << 32) | (unsigned)(~i);
        }
        skeys[i] = k;
    }
    __syncthreads();
    for (int kk = 2; kk <= NP_SORT; kk <<= 1) {
        for (int j = kk >> 1; j > 0; j >>= 1) {
            for (int i = t; i < NP_SORT; i += 1024) {
                int ixj = i ^ j;
                if (ixj > i) {
                    unsigned long long a = skeys[i], b = skeys[ixj];
                    bool desc = ((i & kk) == 0);
                    if ((a < b) == desc) { skeys[i] = b; skeys[ixj] = a; }
                }
            }
            __syncthreads();
        }
    }
    unsigned long long k = skeys[t];
    int idx = (int)(~(unsigned)(k & 0xffffffffu));
    sc_out[t] = scores[idx];
    cl_out[t] = classes[idx];
    b_out[t * 4 + 0] = boxes[idx * 4 + 0];
    b_out[t * 4 + 1] = boxes[idx * 4 + 1];
    b_out[t * 4 + 2] = boxes[idx * 4 + 2];
    b_out[t * 4 + 3] = boxes[idx * 4 + 3];
}

// ---------------------------------------------------------------------------
// K3: greedy NMS (exact scan semantics) + allowed-class/score mask
// ---------------------------------------------------------------------------
__global__ __launch_bounds__(1024) void k_nms(const float* __restrict__ b,
                                              const float* __restrict__ sc,
                                              const int*   __restrict__ cl,
                                              const int*   __restrict__ allowed,
                                              int n_allowed,
                                              float* __restrict__ masked) {
    __shared__ float sx1[PRE_NMS], sy1[PRE_NMS], sx2[PRE_NMS], sy2[PRE_NMS];
    __shared__ int   ssup[PRE_NMS];
    int j = threadIdx.x;
    float x1 = b[j * 4 + 0], y1 = b[j * 4 + 1];
    float x2 = b[j * 4 + 2], y2 = b[j * 4 + 3];
    sx1[j] = x1; sy1[j] = y1; sx2[j] = x2; sy2[j] = y2;
    ssup[j] = 0;
    float areaj = (x2 - x1) * (y2 - y1);
    __syncthreads();
    for (int i = 0; i < PRE_NMS - 1; ++i) {
        if (j > i && ssup[i] == 0) {
            float ax1 = sx1[i], ay1 = sy1[i], ax2 = sx2[i], ay2 = sy2[i];
            float lx = fmaxf(ax1, x1), ly = fmaxf(ay1, y1);
            float rx = fminf(ax2, x2), ry = fminf(ay2, y2);
            float iw = fmaxf(rx - lx, 0.0f), ih = fmaxf(ry - ly, 0.0f);
            float inter = iw * ih;
            float areai = (ax2 - ax1) * (ay2 - ay1);
            float uni   = fmaxf(areai + areaj - inter, 1e-12f);
            if (inter / uni > 0.45f) ssup[j] = 1;
        }
        __syncthreads();
    }
    bool keep = (ssup[j] == 0);
    int  c = cl[j];
    bool in_allowed = false;
    for (int q = 0; q < n_allowed; ++q) in_allowed |= (c == allowed[q]);
    float s = sc[j];
    masked[j] = (keep && (s > 0.1f) && in_allowed) ? s : -__builtin_inff();
}

// ---------------------------------------------------------------------------
// K4: top-31 of masked (bitonic 1024 in LDS), build final 32 rows, emit
//     96 scaled+clipped crop boxes
// ---------------------------------------------------------------------------
__global__ __launch_bounds__(1024) void k_final(const float* __restrict__ b,
                                                const float* __restrict__ sc,
                                                const int*   __restrict__ cl,
                                                const float* __restrict__ masked,
                                                float* __restrict__ out,      // d_out head
                                                float* __restrict__ scaled) { // 96*4
    __shared__ unsigned long long skeys[PRE_NMS];
    __shared__ float fb[32][4];
    int t = threadIdx.x;
    {
        unsigned u = float_key(masked[t]);
        skeys[t] = ((unsigned long long)u << 32) | (unsigned)(~t);
    }
    __syncthreads();
    for (int kk = 2; kk <= PRE_NMS; kk <<= 1) {
        for (int j = kk >> 1; j > 0; j >>= 1) {
            int ixj = t ^ j;
            if (ixj > t) {
                unsigned long long a = skeys[t], bb = skeys[ixj];
                bool desc = ((t & kk) == 0);
                if ((a < bb) == desc) { skeys[t] = bb; skeys[ixj] = a; }
            }
            __syncthreads();
        }
    }
    if (t < M_MAX) {
        unsigned long long k = skeys[t];
        int idx = (int)(~(unsigned)(k & 0xffffffffu));
        float v = masked[idx];
        bool valid = (v > -__builtin_inff());
        float bx1 = valid ? b[idx * 4 + 0] : 0.0f;
        float by1 = valid ? b[idx * 4 + 1] : 0.0f;
        float bx2 = valid ? b[idx * 4 + 2] : 0.0f;
        float by2 = valid ? b[idx * 4 + 3] : 0.0f;
        float s   = valid ? sc[idx] : 0.0f;
        float c   = valid ? (float)cl[idx] : 0.0f;
        int r = t + 1;
        out[r * 4 + 0] = bx1; out[r * 4 + 1] = by1;
        out[r * 4 + 2] = bx2; out[r * 4 + 3] = by2;
        out[128 + r] = s;
        out[160 + r] = c;
        fb[r][0] = bx1; fb[r][1] = by1; fb[r][2] = bx2; fb[r][3] = by2;
    }
    if (t == M_MAX) {   // row 0: full-image box
        out[0] = 0.0f; out[1] = 0.0f; out[2] = (float)(W_IMG - 1); out[3] = (float)(H_IMG - 1);
        out[128] = 1.0f;
        out[160] = -1.0f;
        fb[0][0] = 0.0f; fb[0][1] = 0.0f; fb[0][2] = (float)(W_IMG - 1); fb[0][3] = (float)(H_IMG - 1);
    }
    __syncthreads();
    if (t < 96) {
        int r = t / 3, si = t % 3;
        float scale = (si == 0) ? 0.7071f : ((si == 1) ? 1.0f : 1.4142f);
        float bx1 = fb[r][0], by1 = fb[r][1], bx2 = fb[r][2], by2 = fb[r][3];
        float ccx = (bx1 + bx2) * 0.5f, ccy = (by1 + by2) * 0.5f;
        float ww = (bx2 - bx1) * scale * 0.5f;
        float hh = (by2 - by1) * scale * 0.5f;
        scaled[t * 4 + 0] = fminf(fmaxf(ccx - ww, 0.0f), (float)(W_IMG - 1));
        scaled[t * 4 + 1] = fminf(fmaxf(ccy - hh, 0.0f), (float)(H_IMG - 1));
        scaled[t * 4 + 2] = fminf(fmaxf(ccx + ww, 0.0f), (float)(W_IMG - 1));
        scaled[t * 4 + 3] = fminf(fmaxf(ccy + hh, 0.0f), (float)(H_IMG - 1));
    }
}

// ---------------------------------------------------------------------------
// K5: bilinear crop + normalize via WMMA.
//   One wave = one 16x16 output tile of one crop, ALL 3 channels:
//   the interpolation geometry (row params + sparse B weights) is channel-
//   invariant, so the B fragments (both K-chunks) are built once and feed
//   three accumulator chains -> 6 v_wmma_f32_16x16x32_f16 per wave.
//   Window of 64 source cols always covers the 16-col tile (bw<=639/224).
//   EXEC all-ones: grid == exact tile count.
// ---------------------------------------------------------------------------
__global__ __launch_bounds__(256) void k_crops(const float* __restrict__ img,
                                               const float* __restrict__ scaled,
                                               float* __restrict__ out) {
    __shared__ _Float16 tile[8][3][16][72];   // per-wave slice, padded
    int lane = threadIdx.x & 31;
    int wv   = threadIdx.x >> 5;
    int tid  = blockIdx.x * 8 + wv;

    int tin  = tid % 196;          // 14x14 tiles per crop
    int crop = tid / 196;
    int r0 = (tin / 14) * 16;
    int c0 = (tin % 14) * 16;

    float bx1 = scaled[crop * 4 + 0], by1 = scaled[crop * 4 + 1];
    float bx2 = scaled[crop * 4 + 2], by2 = scaled[crop * 4 + 3];
    float bw = (bx2 - bx1) * (1.0f / OUT_SZ);
    float bh = (by2 - by1) * (1.0f / OUT_SZ);

    float xs0  = bx1 - 0.5f + ((float)c0 + 0.5f) * bw;
    float xsc0 = fminf(fmaxf(xs0, 0.0f), (float)(W_IMG - 1));
    int   xbase = (int)floorf(xsc0);

    // vertical blend into LDS: 16 rows x 64 cols, row params shared by 3 ch
    for (int it = 0; it < 32; ++it) {
        int e    = lane + it * 32;
        int rr   = e >> 6;
        int colk = e & 63;
        float ysf = by1 - 0.5f + ((float)(r0 + rr) + 0.5f) * bh;
        float myf = (ysf > -1.0f && ysf < (float)H_IMG) ? 1.0f : 0.0f;
        float ysc = fminf(fmaxf(ysf, 0.0f), (float)(H_IMG - 1));
        int   y0  = (int)floorf(ysc);
        int   y1i = (y0 + 1 < H_IMG - 1) ? (y0 + 1) : (H_IMG - 1);
        float wy  = ysc - (float)y0;
        int   xsi = xbase + colk;
        if (xsi > W_IMG - 1) xsi = W_IMG - 1;
        #pragma unroll
        for (int ch = 0; ch < 3; ++ch) {
            const float* imgc = img + (size_t)ch * H_IMG * W_IMG;
            float v = imgc[y0 * W_IMG + xsi] * (1.0f - wy) + imgc[y1i * W_IMG + xsi] * wy;
            tile[wv][ch][rr][colk] = (_Float16)(v * myf);
        }
    }
    __syncthreads();

    // per-lane horizontal weights (col mask folded in) -- channel-invariant
    int nloc = lane & 15, half = lane >> 4;
    int c = c0 + nloc;
    float xsf = bx1 - 0.5f + ((float)c + 0.5f) * bw;
    float mxf = (xsf > -1.0f && xsf < (float)W_IMG) ? 1.0f : 0.0f;
    float xsc = fminf(fmaxf(xsf, 0.0f), (float)(W_IMG - 1));
    int   x0  = (int)floorf(xsc);
    float wx  = xsc - (float)x0;
    float w0  = (1.0f - wx) * mxf;
    float w1  = wx * mxf;
    int p0 = x0 - xbase, p1 = p0 + 1;

    // build both B fragments once
    v16h bf0, bf1;
    #pragma unroll
    for (int v = 0; v < 8; ++v) {
        int k = 8 * half + 2 * v + ((v >= 4) ? 8 : 0);
        bf0[2 * v]     = (_Float16)((k == p0) ? w0 : ((k == p1) ? w1 : 0.0f));
        bf0[2 * v + 1] = (_Float16)(((k + 1) == p0) ? w0 : (((k + 1) == p1) ? w1 : 0.0f));
        int k2 = 32 + k;
        bf1[2 * v]     = (_Float16)((k2 == p0) ? w0 : ((k2 == p1) ? w1 : 0.0f));
        bf1[2 * v + 1] = (_Float16)(((k2 + 1) == p0) ? w0 : (((k2 + 1) == p1) ? w1 : 0.0f));
    }

    int col = c0 + nloc;
    #pragma unroll
    for (int ch = 0; ch < 3; ++ch) {
        v8f acc = {};
        #pragma unroll
        for (int chunk = 0; chunk < 2; ++chunk) {
            int kc = chunk * 32;
            v16h a;
            // A fragment: row M = lane%16; K subsets per wave32 16-bit layout
            #pragma unroll
            for (int v = 0; v < 4; ++v) {
                int k = kc + 8 * half + 2 * v;
                a[2 * v]     = tile[wv][ch][nloc][k];
                a[2 * v + 1] = tile[wv][ch][nloc][k + 1];
            }
            #pragma unroll
            for (int v = 4; v < 8; ++v) {
                int k = kc + 16 + 8 * half + 2 * (v - 4);
                a[2 * v]     = tile[wv][ch][nloc][k];
                a[2 * v + 1] = tile[wv][ch][nloc][k + 1];
            }
            acc = __builtin_amdgcn_wmma_f32_16x16x32_f16(false, a, false,
                                                         (chunk == 0) ? bf0 : bf1,
                                                         (short)0, acc, false, false);
        }
        float mu  = (ch == 0) ? 0.485f : ((ch == 1) ? 0.456f : 0.406f);
        float inv = (ch == 0) ? (1.0f / 0.229f) : ((ch == 1) ? (1.0f / 0.224f) : (1.0f / 0.225f));
        float* outp = out + ((size_t)(crop * 3 + ch) * OUT_SZ) * OUT_SZ;
        #pragma unroll
        for (int j = 0; j < 8; ++j) {
            int row = r0 + j + 8 * half;   // D layout: VGPR j -> M = j + 8*half
            outp[row * OUT_SZ + col] = (acc[j] - mu) * inv;
        }
    }
}

// ---------------------------------------------------------------------------
extern "C" void kernel_launch(void* const* d_in, const int* in_sizes, int n_in,
                              void* d_out, int out_size, void* d_ws, size_t ws_size,
                              hipStream_t stream) {
    const float* raw     = (const float*)d_in[0];   // (1,604,8400)
    const float* image   = (const float*)d_in[1];   // (1,3,640,640)
    const int*   allowed = (const int*)d_in[2];
    int n_allowed = in_sizes[2];
    float* out = (float*)d_out;

    float* ws       = (float*)d_ws;
    float* scores   = ws;                         // 8400
    int*   classes  = (int*)(ws + 8400);          // 8400
    float* boxes    = ws + 16800;                 // 8400*4
    float* b_s      = ws + 50400;                 // 1024*4
    float* sc_s     = ws + 54496;                 // 1024
    int*   cl_s     = (int*)(ws + 55520);         // 1024
    float* masked   = ws + 56544;                 // 1024
    float* scaledbx = ws + 57568;                 // 96*4

    k_decode<<<(N_ANCH / 4 + 255) / 256, 256, 0, stream>>>(raw, scores, classes, boxes);
    k_sort_topk<<<1, 1024, NP_SORT * sizeof(unsigned long long), stream>>>(
        scores, classes, boxes, b_s, sc_s, cl_s);
    k_nms<<<1, 1024, 0, stream>>>(b_s, sc_s, cl_s, allowed, n_allowed, masked);
    k_final<<<1, 1024, 0, stream>>>(b_s, sc_s, cl_s, masked, out, scaledbx);
    // 96 crops * 196 tiles = 18816 waves; 8 waves/block -> 2352 blocks
    k_crops<<<2352, 256, 0, stream>>>(image, scaledbx, out + 192);
}
